// GCNLink_18743237280603
// MI455X (gfx1250) — matrix operand order repaired and checked by reading, not compiled, not run on hardware
//
#include <hip/hip_runtime.h>
#include <hip/hip_bf16.h>
#include <math.h>

#define IN_DIM 256
#define HID    128

typedef __attribute__((ext_vector_type(16))) __bf16 v16bf;
typedef __attribute__((ext_vector_type(8)))  float  v8f;

__device__ __forceinline__ void atomic_add_f32(float* p, float v) {
    (void)__hip_atomic_fetch_add(p, v, __ATOMIC_RELAXED, __HIP_MEMORY_SCOPE_AGENT);
}

// ---------------------------------------------------------------------------
// COO SpMM scatter: out[row[e], :] += val[e] * x[col[e], :]
// One wave per non-zero; lanes stride the feature dim -> coalesced 128B reads
// of the gathered row (L2-resident: x=102MB, h1=51MB vs 192MB L2).
// ---------------------------------------------------------------------------
template<int D>
__global__ __launch_bounds__(256)
void spmm_scatter(const int* __restrict__ arow, const int* __restrict__ acol,
                  const float* __restrict__ aval, const float* __restrict__ x,
                  float* __restrict__ out, int nnz)
{
    const int lane = threadIdx.x & 31;
    const int e = (int)(((size_t)blockIdx.x * blockDim.x + threadIdx.x) >> 5);
    if (e >= nnz) return;
    const int   r = arow[e];
    const int   c = acol[e];
    const float v = aval[e];
    const float* __restrict__ xr   = x   + (size_t)c * D;
    float*       __restrict__ orow = out + (size_t)r * D;
#pragma unroll
    for (int d = lane; d < D; d += 32)
        atomic_add_f32(&orow[d], v * xr[d]);
}

// ---------------------------------------------------------------------------
// out[M, HID] = relu( A[M, KDIM] @ W[HID, KDIM]^T )  via bf16 WMMA, f32 accum.
// One wave per 16x16 output tile; 8 waves/block cover the 8 N-tiles of HID.
// Operand layouts per CDNA5 ISA 7.12.2 (wave32; 16-bit A 16x32, B 32x16):
//   A lane L: row M=L&15; elems 0-7 -> K=kb+8*(L>>4)+j, elems 8-15 -> +16
//   B lane L: col N=L&15; elems 0-7 -> K=kb+8*(L>>4)+j, elems 8-15 -> +16
//   D VGPR v: row M=v+8*(L>>4), col N=L&15
// ---------------------------------------------------------------------------
template<int KDIM>
__global__ __launch_bounds__(256)
void gemm_wt_relu_wmma(const float* __restrict__ A,  // [M, KDIM]
                       const float* __restrict__ W,  // [HID, KDIM]
                       float* __restrict__ out, int M)
{
    const int lane = threadIdx.x & 31;
    const int wave = threadIdx.x >> 5;     // tileN index 0..7
    const int half = lane >> 4;
    const int l15  = lane & 15;
    const int am   = blockIdx.x * 16 + l15;
    const int amc  = am < M ? am : (M - 1);          // clamp A reads
    const int bn   = wave * 16 + l15;

    // Segment bases: A 32B-aligned, B 32B-aligned -> b128 loads are legal.
    const float4* __restrict__ ap = (const float4*)(A + (size_t)amc * KDIM + 8  * half);
    const float4* __restrict__ bp = (const float4*)(W + (size_t)bn  * KDIM + 8  * half);

    v8f c = {};
#pragma unroll 2
    for (int kb = 0; kb < KDIM; kb += 32) {
        const int q = kb >> 2;                       // float4 index of kb
        float4 a0 = ap[q + 0], a1 = ap[q + 1];       // K = kb+8h .. +7
        float4 a2 = ap[q + 4], a3 = ap[q + 5];       // K = kb+16+8h .. +7
        float4 b0 = bp[q + 0], b1 = bp[q + 1];       // K = kb+8h .. +7
        float4 b2 = bp[q + 4], b3 = bp[q + 5];       // K = kb+16+8h .. +7
        v16bf a, b;
        a[0]=(__bf16)a0.x; a[1]=(__bf16)a0.y; a[2]=(__bf16)a0.z; a[3]=(__bf16)a0.w;
        a[4]=(__bf16)a1.x; a[5]=(__bf16)a1.y; a[6]=(__bf16)a1.z; a[7]=(__bf16)a1.w;
        a[8]=(__bf16)a2.x; a[9]=(__bf16)a2.y; a[10]=(__bf16)a2.z; a[11]=(__bf16)a2.w;
        a[12]=(__bf16)a3.x; a[13]=(__bf16)a3.y; a[14]=(__bf16)a3.z; a[15]=(__bf16)a3.w;
        b[0]=(__bf16)b0.x; b[1]=(__bf16)b0.y; b[2]=(__bf16)b0.z; b[3]=(__bf16)b0.w;
        b[4]=(__bf16)b1.x; b[5]=(__bf16)b1.y; b[6]=(__bf16)b1.z; b[7]=(__bf16)b1.w;
        b[8]=(__bf16)b2.x; b[9]=(__bf16)b2.y; b[10]=(__bf16)b2.z; b[11]=(__bf16)b2.w;
        b[12]=(__bf16)b3.x; b[13]=(__bf16)b3.y; b[14]=(__bf16)b3.z; b[15]=(__bf16)b3.w;
        // (neg_a, A, neg_b, B, c_mod, C, reuse_a, reuse_b)
        c = __builtin_amdgcn_wmma_f32_16x16x32_bf16(false, a, false, b,
                                                    (short)0, c, false, false);
    }

    const int mbase = blockIdx.x * 16 + 8 * half;    // D row base for this lane
    float* __restrict__ op = out + (size_t)mbase * HID + wave * 16 + l15;
    if (mbase + 7 < M) {
        // Fast path (always taken when M % 16 == 0): no per-element guards.
#pragma unroll
        for (int v = 0; v < 8; ++v) {
            float r = c[v];
            op[(size_t)v * HID] = r > 0.f ? r : 0.f;
        }
    } else {
#pragma unroll
        for (int v = 0; v < 8; ++v) {
            if (mbase + v < M) {
                float r = c[v];
                op[(size_t)v * HID] = r > 0.f ? r : 0.f;
            }
        }
    }
}

// ---------------------------------------------------------------------------
// Link scoring: sigmoid( [h[src] ; h[dst]] . Wl + bl )
// One thread per edge; float4 loads (512B contiguous per gathered row, L2 hit).
// Wl/bl are thread-uniform -> scalar loads.
// ---------------------------------------------------------------------------
__global__ __launch_bounds__(256)
void edge_score(const float* __restrict__ h, const int* __restrict__ ei,
                const float* __restrict__ Wl, const float* __restrict__ bl,
                float* __restrict__ out, int E)
{
    const int e = blockIdx.x * blockDim.x + threadIdx.x;
    if (e >= E) return;
    const int s = ei[e];
    const int d = ei[E + e];
    const float4* __restrict__ hs = (const float4*)(h + (size_t)s * HID);
    const float4* __restrict__ hd = (const float4*)(h + (size_t)d * HID);
    const float4* __restrict__ w0 = (const float4*)Wl;
    const float4* __restrict__ w1 = (const float4*)(Wl + HID);
    float acc = bl[0];
#pragma unroll
    for (int k = 0; k < HID / 4; ++k) {
        float4 a = hs[k], wa = w0[k];
        float4 b = hd[k], wb = w1[k];
        acc += a.x * wa.x + a.y * wa.y + a.z * wa.z + a.w * wa.w;
        acc += b.x * wb.x + b.y * wb.y + b.z * wb.z + b.w * wb.w;
    }
    out[e] = 1.0f / (1.0f + __expf(-acc));
}

// ---------------------------------------------------------------------------
extern "C" void kernel_launch(void* const* d_in, const int* in_sizes, int n_in,
                              void* d_out, int out_size, void* d_ws, size_t ws_size,
                              hipStream_t stream)
{
    const float* x          = (const float*)d_in[0];
    const int*   adj_row    = (const int*)  d_in[1];
    const int*   adj_col    = (const int*)  d_in[2];
    const float* adj_val    = (const float*)d_in[3];
    const int*   edge_index = (const int*)  d_in[4];
    const float* W1         = (const float*)d_in[5];  // [HID, IN_DIM]
    const float* W2         = (const float*)d_in[6];  // [HID, HID]
    const float* Wl         = (const float*)d_in[7];  // [1, 2*HID]
    const float* bl         = (const float*)d_in[8];  // [1]

    const int n   = in_sizes[0] / IN_DIM;  // nodes
    const int nnz = in_sizes[1];
    const int E   = in_sizes[4] / 2;

    // Workspace layout (footprint n*(256+128)*4 = 153.6 MB):
    //   [0,     n*256) : h_spmm1  (dead after GEMM1)
    //     [0,     n*128)  : h_spmm2 (aliases lower half)
    //     [n*128, n*256)  : h2      (aliases upper half)
    //   [n*256, n*384) : h1
    float* ws  = (float*)d_ws;
    float* hs1 = ws;
    float* h1  = ws + (size_t)n * IN_DIM;
    float* hs2 = ws;
    float* h2  = ws + (size_t)n * HID;

    const int spmm_blocks = (nnz + 7) / 8;   // 8 waves (= 8 nnz) per 256-thread block
    const int gemm_blocks = (n + 15) / 16;

    // Layer 1: spmm(x) -> [n,256]; relu(. @ W1^T) -> h1 [n,128]
    hipMemsetAsync(hs1, 0, (size_t)n * IN_DIM * sizeof(float), stream);
    spmm_scatter<IN_DIM><<<spmm_blocks, 256, 0, stream>>>(adj_row, adj_col, adj_val,
                                                          x, hs1, nnz);
    gemm_wt_relu_wmma<IN_DIM><<<gemm_blocks, 256, 0, stream>>>(hs1, W1, h1, n);

    // Layer 2: spmm(h1) -> [n,128]; relu(. @ W2^T) -> h2 [n,128]
    hipMemsetAsync(hs2, 0, (size_t)n * HID * sizeof(float), stream);
    spmm_scatter<HID><<<spmm_blocks, 256, 0, stream>>>(adj_row, adj_col, adj_val,
                                                       h1, hs2, nnz);
    gemm_wt_relu_wmma<HID><<<gemm_blocks, 256, 0, stream>>>(hs2, W2, h2, n);

    // Edge link prediction -> sigmoid logits [E]
    edge_score<<<(E + 255) / 256, 256, 0, stream>>>(h2, edge_index, Wl, bl,
                                                    (float*)d_out, E);
}
